// TSP_Decoder_22162031247508
// MI455X (gfx1250) — matrix operand order repaired and checked by reading, not compiled, not run on hardware
//
#include <hip/hip_runtime.h>
#include <hip/hip_bf16.h>

typedef __attribute__((ext_vector_type(16))) _Float16 v16h;
typedef __attribute__((ext_vector_type(8)))  _Float16 v8h;
typedef __attribute__((ext_vector_type(8)))  float    v8f;

// Exact pointee type expected by __builtin_amdgcn_global_load_async_to_lds_b128
// (per hipcc diagnostic: '__attribute__((__vector_size__(4 * sizeof(int)))) int __device__ *').
typedef __attribute__((__vector_size__(4 * sizeof(int)))) int vsi4;
typedef __attribute__((address_space(1))) vsi4* gas_v4i;
typedef __attribute__((address_space(3))) vsi4* las_v4i;

#define SEQ 512
#define EMBD 128
#define NBATCH 32

#define WMMA16(A, B, C) __builtin_amdgcn_wmma_f32_16x16x32_f16(false, (A), false, (B), (short)0, (C), false, false)

// ---------------------------------------------------------------------------
// Load a 16xK A-fragment K-chunk from an fp32 row (row-major, stride EMBD),
// converting to f16.  Lane holds M=lane&15, halves 0-15 = K kb..kb+15,
// where kb = 32*chunk + (lane>=16 ? 16 : 0).
__device__ __forceinline__ v16h loadA_f32(const float* __restrict__ rowp, int kb) {
  float tmp[16];
  *(float4*)&tmp[0]  = *(const float4*)(rowp + kb);
  *(float4*)&tmp[4]  = *(const float4*)(rowp + kb + 4);
  *(float4*)&tmp[8]  = *(const float4*)(rowp + kb + 8);
  *(float4*)&tmp[12] = *(const float4*)(rowp + kb + 12);
  union { v16h v; _Float16 h[16]; } u;
#pragma unroll
  for (int i = 0; i < 16; ++i) u.h[i] = (_Float16)tmp[i];
  return u.v;
}

// ---------------------------------------------------------------------------
// Kernel 0: transpose + convert the 10 weight matrices (128x128, in-major fp32)
// into out-major f16 so B-fragments are contiguous 32B per-lane loads.
__global__ void prep_weights(const float* Wk, const float* Wv, const float* Wke,
                             const float* Wve, const float* Wqf, const float* Wql,
                             const float* Wqfe, const float* Wqle,
                             const float* Wc1, const float* Wc2,
                             _Float16* __restrict__ WT) {
  const int mat = blockIdx.x >> 7;
  const int o   = blockIdx.x & 127;
  const int i   = threadIdx.x;
  const float* srcs[10] = {Wk, Wv, Wke, Wve, Wqf, Wql, Wqfe, Wqle, Wc1, Wc2};
  WT[mat * 16384 + o * 128 + i] = (_Float16)srcs[mat][i * 128 + o];
}

// ---------------------------------------------------------------------------
// Kernel 1: projections.  1024 blocks (= B * 32 row-tiles), 8 waves/block.
// Wave w computes the 16-column tile [16w,16w+16) of K, V, Ke, Ve and the
// 4-way-summed Q for its 16-row tile.  V/Ve stored transposed (B,128,512).
__global__ __launch_bounds__(256)
void proj_kernel(const float* __restrict__ nodes, const float* __restrict__ graph,
                 const float* __restrict__ q1, const float* __restrict__ fg,
                 const float* __restrict__ ln, const float* __restrict__ lg,
                 const _Float16* __restrict__ WT,
                 _Float16* __restrict__ Q,  _Float16* __restrict__ Kn,
                 _Float16* __restrict__ Ke, _Float16* __restrict__ Vt,
                 _Float16* __restrict__ Vet, _Float16* __restrict__ Shk) {
  const int b    = blockIdx.x >> 5;
  const int n0   = (blockIdx.x & 31) << 4;
  const int tid  = threadIdx.x;
  const int w    = tid >> 5;
  const int lane = tid & 31;
  const int col  = lane & 15;
  const int hi   = lane >> 4;

  // shk = nodes + graph (f16), 8 elements per thread, vectorized.
  {
    const int e0  = tid * 8;
    const int row = e0 >> 7, c = e0 & 127;
    const size_t base = ((size_t)(b * SEQ + n0 + row)) * EMBD + c;
    float4 a0 = *(const float4*)(nodes + base);
    float4 a1 = *(const float4*)(nodes + base + 4);
    float4 g0 = *(const float4*)(graph + base);
    float4 g1 = *(const float4*)(graph + base + 4);
    union { v8h v; _Float16 h[8]; } us;
    us.h[0] = (_Float16)(a0.x + g0.x); us.h[1] = (_Float16)(a0.y + g0.y);
    us.h[2] = (_Float16)(a0.z + g0.z); us.h[3] = (_Float16)(a0.w + g0.w);
    us.h[4] = (_Float16)(a1.x + g1.x); us.h[5] = (_Float16)(a1.y + g1.y);
    us.h[6] = (_Float16)(a1.z + g1.z); us.h[7] = (_Float16)(a1.w + g1.w);
    *(v8h*)(Shk + base) = us.v;
  }

  const int    arow  = n0 + col;                      // A-matrix row for this lane
  const size_t rbase = ((size_t)(b * SEQ + arow)) * EMBD;
  const _Float16* WkT   = WT + 0 * 16384;
  const _Float16* WvT   = WT + 1 * 16384;
  const _Float16* WkeT  = WT + 2 * 16384;
  const _Float16* WveT  = WT + 3 * 16384;
  const _Float16* WqfT  = WT + 4 * 16384;
  const _Float16* WqlT  = WT + 5 * 16384;
  const _Float16* WqfeT = WT + 6 * 16384;
  const _Float16* WqleT = WT + 7 * 16384;

  v8f kacc = {}, vacc = {}, keacc = {}, veacc = {}, qacc = {};
#pragma unroll
  for (int c = 0; c < 4; ++c) {
    const int kb = 32 * c + hi * 16;                  // A-frag K window
    const int ob = (16 * w + col) * EMBD + 32 * c + hi * 16; // B-frag offset
    v16h an = loadA_f32(nodes + rbase, kb);
    v16h ag = loadA_f32(graph + rbase, kb);
    kacc  = WMMA16(an, *(const v16h*)(WkT  + ob), kacc);
    vacc  = WMMA16(an, *(const v16h*)(WvT  + ob), vacc);
    keacc = WMMA16(ag, *(const v16h*)(WkeT + ob), keacc);
    veacc = WMMA16(ag, *(const v16h*)(WveT + ob), veacc);
    qacc  = WMMA16(loadA_f32(q1 + rbase, kb), *(const v16h*)(WqfT  + ob), qacc);
    qacc  = WMMA16(loadA_f32(ln + rbase, kb), *(const v16h*)(WqlT  + ob), qacc);
    qacc  = WMMA16(loadA_f32(fg + rbase, kb), *(const v16h*)(WqfeT + ob), qacc);
    qacc  = WMMA16(loadA_f32(lg + rbase, kb), *(const v16h*)(WqleT + ob), qacc);
  }

  // C/D layout: value[r] lives at row r + 8*hi, column 16w + col.
  const int    half8 = hi * 8;
  const size_t cbase = ((size_t)(b * SEQ + n0 + half8)) * EMBD + 16 * w + col;
#pragma unroll
  for (int r = 0; r < 8; ++r) {
    Kn[cbase + (size_t)r * EMBD] = (_Float16)kacc[r];
    Ke[cbase + (size_t)r * EMBD] = (_Float16)keacc[r];
    Q [cbase + (size_t)r * EMBD] = (_Float16)qacc[r];
  }
  // Transposed V stores: 8 consecutive m per lane -> single 16B store.
  union { v8h v; _Float16 h[8]; } uv, uve;
#pragma unroll
  for (int r = 0; r < 8; ++r) { uv.h[r] = (_Float16)vacc[r]; uve.h[r] = (_Float16)veacc[r]; }
  const size_t tbase = ((size_t)(b * EMBD + 16 * w + col)) * SEQ + n0 + half8;
  *(v8h*)(Vt  + tbase) = uv.v;
  *(v8h*)(Vet + tbase) = uve.v;
}

// ---------------------------------------------------------------------------
// Fused per-head attention (softmax over all 512 keys, nothing spilled to HBM).
// Processes 32 keys per iteration: two d=16-padded score WMMAs feed one
// full-K=32 P@V WMMA.  Score B-frag upper lanes carry (finite) garbage that is
// annihilated by the zeroed upper half of the Q A-fragment, so no lane
// predication is needed anywhere in the hot loop.
__device__ __forceinline__ void attn_one(const _Float16* __restrict__ K,
                                         const _Float16* __restrict__ Vt,
                                         const float (*mask)[SEQ],
                                         _Float16 (*Pb)[32],
                                         _Float16 (*O)[EMBD],
                                         v16h qa, int b, int h, int col, int hi) {
  const int half8 = hi * 8;
  v8f   o = {};
  float sacc[8] = {0, 0, 0, 0, 0, 0, 0, 0};
  const size_t kbase = (size_t)b * SEQ * EMBD + 16 * h;
  const size_t vbase = ((size_t)(b * EMBD + 16 * h + col)) * SEQ + hi * 16;

  for (int mt = 0; mt < 16; ++mt) {
    const int m0 = mt << 5;
    v16h kf0 = *(const v16h*)(K + kbase + (size_t)(m0 + col) * EMBD);
    v16h kf1 = *(const v16h*)(K + kbase + (size_t)(m0 + 16 + col) * EMBD);
    if (mt < 15)
      __builtin_prefetch(K + kbase + (size_t)(m0 + 32 + col) * EMBD, 0, 1);
    v8f zc = {};
    v8f s0 = WMMA16(qa, kf0, zc);
    v8f s1 = WMMA16(qa, kf1, zc);
#pragma unroll
    for (int r = 0; r < 8; ++r) {
      float e0 = __expf(s0[r] * 0.25f + mask[r + half8][m0 + col]);       // 1/sqrt(16)
      float e1 = __expf(s1[r] * 0.25f + mask[r + half8][m0 + 16 + col]);
      sacc[r] += e0 + e1;
      Pb[r + half8][col]      = (_Float16)e0;   // C-layout -> LDS tile
      Pb[r + half8][16 + col] = (_Float16)e1;
    }
    // Re-read P as a full-K=32 A-fragment (row = col).
    union { v16h v; v8h p[2]; } ua;
    ua.p[0] = *(const v8h*)&Pb[col][hi * 16];
    ua.p[1] = *(const v8h*)&Pb[col][hi * 16 + 8];
    // B-frag of V from transposed layout: all 32 lanes load 16 contiguous m.
    v16h vf = *(const v16h*)(Vt + vbase + m0);
    o = WMMA16(ua.v, vf, o);
  }
  // Row sums: reduce over the 16-lane group, normalize, park head slice in LDS.
#pragma unroll
  for (int r = 0; r < 8; ++r) {
    float t = sacc[r];
    t += __shfl_xor(t, 1, 32);
    t += __shfl_xor(t, 2, 32);
    t += __shfl_xor(t, 4, 32);
    t += __shfl_xor(t, 8, 32);
    O[r + half8][16 * h + col] = (_Float16)(o[r] * (1.0f / t));
  }
}

// ---------------------------------------------------------------------------
// Kernel 2: fused attention x2 + combine + node scoring + final softmax.
// 1024 blocks (= B * 32 query tiles), 8 waves; wave = head (phase A),
// column tile (phase B), key range (phase C).
__global__ __launch_bounds__(256)
void attn_kernel(const _Float16* __restrict__ Q,  const _Float16* __restrict__ Kn,
                 const _Float16* __restrict__ Ke, const _Float16* __restrict__ Vt,
                 const _Float16* __restrict__ Vet, const _Float16* __restrict__ Shk,
                 const _Float16* __restrict__ WT, const float* __restrict__ mask,
                 const float* __restrict__ b1, const float* __restrict__ b2,
                 float* __restrict__ out) {
  __shared__ float    lds_mask[16][SEQ];     // 32 KB; reused as exp() storage in phase C
  __shared__ _Float16 lds_O1[16][EMBD];      // 4 KB
  __shared__ _Float16 lds_O2[16][EMBD];      // 4 KB
  __shared__ _Float16 lds_mh[16][EMBD];      // 4 KB
  __shared__ _Float16 lds_P[8][16][32];      // 8 KB (per-wave P tiles, 32 keys)
  __shared__ float    lds_rowsum[8][16];

  const int b    = blockIdx.x >> 5;
  const int n0   = (blockIdx.x & 31) << 4;
  const int tid  = threadIdx.x;
  const int w    = tid >> 5;
  const int lane = tid & 31;
  const int col  = lane & 15;
  const int hi   = lane >> 4;
  const int half8 = hi * 8;

  // Stage the mask slab once (shared by both attentions, all heads, phase C).
#if __has_builtin(__builtin_amdgcn_global_load_async_to_lds_b128)
#pragma unroll
  for (int i = 0; i < 8; ++i) {
    const int idx = (tid + i * 256) * 4;
    const int rr = idx >> 9, m = idx & 511;
    __builtin_amdgcn_global_load_async_to_lds_b128(
        (gas_v4i)(mask + ((size_t)(b * SEQ + n0 + rr)) * SEQ + m),
        (las_v4i)&lds_mask[rr][m], 0, 0);
  }
#if __has_builtin(__builtin_amdgcn_s_wait_asynccnt)
  __builtin_amdgcn_s_wait_asynccnt(0);
#else
  asm volatile("s_wait_asynccnt 0x0" ::: "memory");
#endif
#else
#pragma unroll
  for (int i = 0; i < 8; ++i) {
    const int idx = (tid + i * 256) * 4;
    const int rr = idx >> 9, m = idx & 511;
    *(float4*)&lds_mask[rr][m] =
        *(const float4*)(mask + ((size_t)(b * SEQ + n0 + rr)) * SEQ + m);
  }
#endif
  __syncthreads();

  // ---- Phase A: two attentions, wave = head --------------------------------
  const int h = w;
  v16h qa;
  {
    union { v16h v; v8h p[2]; } u; u.v = (v16h){};
    if (hi == 0) u.v = *(const v16h*)(Q + ((size_t)(b * SEQ + n0 + col)) * EMBD + 16 * h);
    qa = u.v;   // d=16 padded into K=32 (upper lanes zero)
  }
  attn_one(Kn, Vt,  lds_mask, lds_P[w], lds_O1, qa, b, h, col, hi);
  attn_one(Ke, Vet, lds_mask, lds_P[w], lds_O2, qa, b, h, col, hi);
  __syncthreads();

  // ---- Phase B: mh = O1 @ Wc1 + O2 @ Wc2 + (b1 + b2), wave = 16-col tile ---
  const _Float16* Wc1T = WT + 8 * 16384;
  const _Float16* Wc2T = WT + 9 * 16384;
  const float bias = b1[16 * w + col] + b2[16 * w + col];
  v8f mh;
#pragma unroll
  for (int r = 0; r < 8; ++r) mh[r] = bias;
#pragma unroll
  for (int c = 0; c < 4; ++c) {
    const int kb = 32 * c + hi * 16;
    union { v16h v; v8h p[2]; } a1, a2;
    a1.p[0] = *(const v8h*)&lds_O1[col][kb];
    a1.p[1] = *(const v8h*)&lds_O1[col][kb + 8];
    a2.p[0] = *(const v8h*)&lds_O2[col][kb];
    a2.p[1] = *(const v8h*)&lds_O2[col][kb + 8];
    const int ob = (16 * w + col) * EMBD + 32 * c + hi * 16;
    mh = WMMA16(a1.v, *(const v16h*)(Wc1T + ob), mh);
    mh = WMMA16(a2.v, *(const v16h*)(Wc2T + ob), mh);
  }
#pragma unroll
  for (int r = 0; r < 8; ++r) lds_mh[r + half8][16 * w + col] = (_Float16)mh[r];
  __syncthreads();

  // ---- Phase C: score = 10*tanh(mh @ shk^T / sqrt(128)) + mask, softmax ----
  v16h ma[4];
#pragma unroll
  for (int c = 0; c < 4; ++c) {
    const int kb = 32 * c + hi * 16;
    union { v16h v; v8h p[2]; } u;
    u.p[0] = *(const v8h*)&lds_mh[col][kb];
    u.p[1] = *(const v8h*)&lds_mh[col][kb + 8];
    ma[c] = u.v;
  }
  float sacc2[8] = {0, 0, 0, 0, 0, 0, 0, 0};
#pragma unroll
  for (int t = 0; t < 4; ++t) {
    const int m0 = 64 * w + 16 * t;        // this wave's key range
    v8f s2 = {};
#pragma unroll
    for (int c = 0; c < 4; ++c) {
      const v16h bf = *(const v16h*)(Shk + ((size_t)(b * SEQ + m0 + col)) * EMBD +
                                     32 * c + hi * 16);
      s2 = WMMA16(ma[c], bf, s2);
    }
#pragma unroll
    for (int r = 0; r < 8; ++r) {
      float sc = s2[r] * 0.08838834764831845f;            // 1/sqrt(128)
      sc = 10.0f * tanhf(sc) + lds_mask[r + half8][m0 + col];
      float e = __expf(sc);                               // bounded by e^10: stable
      sacc2[r] += e;
      lds_mask[r + half8][m0 + col] = e;                  // overwrite mask in place
    }
  }
#pragma unroll
  for (int r = 0; r < 8; ++r) {
    float t2 = sacc2[r];
    t2 += __shfl_xor(t2, 1, 32);
    t2 += __shfl_xor(t2, 2, 32);
    t2 += __shfl_xor(t2, 4, 32);
    t2 += __shfl_xor(t2, 8, 32);
    sacc2[r] = t2;
  }
  if (col == 0) {
#pragma unroll
    for (int r = 0; r < 8; ++r) lds_rowsum[w][r + half8] = sacc2[r];
  }
  __syncthreads();
  float rinv[8];
#pragma unroll
  for (int r = 0; r < 8; ++r) {
    float tt = 0.0f;
#pragma unroll
    for (int w2 = 0; w2 < 8; ++w2) tt += lds_rowsum[w2][r + half8];
    rinv[r] = 1.0f / tt;
  }
#pragma unroll
  for (int t = 0; t < 4; ++t) {
    const int m0 = 64 * w + 16 * t;
#pragma unroll
    for (int r = 0; r < 8; ++r) {
      out[((size_t)(b * SEQ + n0 + r + half8)) * SEQ + m0 + col] =
          lds_mask[r + half8][m0 + col] * rinv[r];
    }
  }
}

// ---------------------------------------------------------------------------
extern "C" void kernel_launch(void* const* d_in, const int* in_sizes, int n_in,
                              void* d_out, int out_size, void* d_ws, size_t ws_size,
                              hipStream_t stream) {
  (void)in_sizes; (void)n_in; (void)out_size; (void)ws_size;
  const float* nodes = (const float*)d_in[0];
  const float* graph = (const float*)d_in[1];
  const float* q1    = (const float*)d_in[2];
  const float* fg    = (const float*)d_in[3];
  const float* ln    = (const float*)d_in[4];
  const float* lg    = (const float*)d_in[5];
  const float* mask  = (const float*)d_in[6];
  const float* Wqf   = (const float*)d_in[7];
  const float* Wql   = (const float*)d_in[8];
  const float* Wk    = (const float*)d_in[9];
  const float* Wv    = (const float*)d_in[10];
  const float* Wqfe  = (const float*)d_in[11];
  const float* Wqle  = (const float*)d_in[12];
  const float* Wke   = (const float*)d_in[13];
  const float* Wve   = (const float*)d_in[14];
  const float* Wc1   = (const float*)d_in[15];
  const float* bc1   = (const float*)d_in[16];
  const float* Wc2   = (const float*)d_in[17];
  const float* bc2   = (const float*)d_in[18];
  float* out = (float*)d_out;

  // Workspace layout (f16): 10 transposed weights, then 6 (B,512,128) tensors.
  const size_t NT = (size_t)NBATCH * SEQ * EMBD;    // 2M elements = 4 MiB each
  _Float16* WT  = (_Float16*)d_ws;
  _Float16* Qb  = WT + 10 * 16384;
  _Float16* Kn  = Qb + NT;
  _Float16* Ke  = Kn + NT;
  _Float16* Vt  = Ke + NT;
  _Float16* Vet = Vt + NT;
  _Float16* Shk = Vet + NT;

  prep_weights<<<1280, 128, 0, stream>>>(Wk, Wv, Wke, Wve, Wqf, Wql, Wqfe, Wqle,
                                         Wc1, Wc2, WT);
  proj_kernel<<<NBATCH * (SEQ / 16), 256, 0, stream>>>(nodes, graph, q1, fg, ln, lg,
                                                       WT, Qb, Kn, Ke, Vt, Vet, Shk);
  attn_kernel<<<NBATCH * (SEQ / 16), 256, 0, stream>>>(Qb, Kn, Ke, Vt, Vet, Shk, WT,
                                                       mask, bc1, bc2, out);
}
